// Performer_10917806866994
// MI455X (gfx1250) — matrix operand order repaired and checked by reading, not compiled, not run on hardware
//
#include <hip/hip_runtime.h>

typedef __attribute__((ext_vector_type(16))) _Float16 v16h;
typedef __attribute__((ext_vector_type(8)))  _Float16 half8;
typedef __attribute__((ext_vector_type(8)))  float    v8f;

// ---------------- problem constants ----------------
#define BS     16
#define LSEQ   130        // tokens per batch row (131-1)
#define NTOK   (BS*LSEQ)  // 2080
#define DIM    512
#define NHEADS 8
#define DH     64
#define MFEAT  256
#define NROWS  (NTOK*NHEADS)   // 16640 (b,l,h) feature rows
#define VOCAB  523
#define DEPTH  6

// =====================================================================
// Generic fused GEMM:  C = act( A[MxK] @ B + bias + resid )
// A row-major f32, B row-major (KxN) or (NxK if bT), all fp32 in memory,
// f16 WMMA compute with fp32 accumulation.
// block = 128 threads (4 waves), tile 128(M) x 64(N), k-step 32.
// Each wave owns a 32x64 strip: 2 A-frags x 4 B-frags = 8 WMMAs/k-step,
// B-fragments reused from registers across both M-frags.
// =====================================================================
__global__ __launch_bounds__(128) void gemm_wmma_kernel(
    const float* __restrict__ A, int lda,
    const float* __restrict__ B, int ldb,
    const float* __restrict__ bias,
    const float* __restrict__ resid,
    float* __restrict__ C, int ldc,
    int M, int N, int K,
    float scaleA, int bT, int act)
{
    __shared__ __align__(16) _Float16 As[128][40];  // [m][k], padded stride (80B rows)
    __shared__ __align__(16) _Float16 Bsh[64][40];  // [n][k], padded stride

    const int tid  = threadIdx.x;
    const int wave = tid >> 5;
    const int lane = tid & 31;
    const int m0 = blockIdx.y * 128;
    const int n0 = blockIdx.x * 64;

    v8f acc[2][4];
    #pragma unroll
    for (int a = 0; a < 2; a++)
        #pragma unroll
        for (int i = 0; i < 4; i++)
            #pragma unroll
            for (int j = 0; j < 8; j++) acc[a][i][j] = 0.0f;

    const int mrow0 = (wave << 5) + (lane & 15);   // first A-frag row in tile
    const int hi    = (lane >> 4) & 1;

    for (int k0 = 0; k0 < K; k0 += 32) {
        // ---- stage A tile (128x32), convert to f16, scale ----
        for (int i = tid; i < 128 * 32; i += 128) {
            int r = i >> 5, c = i & 31;
            int gr = m0 + r, gc = k0 + c;
            float v = (gr < M) ? A[(size_t)gr * lda + gc] : 0.0f;
            As[r][c] = (_Float16)(v * scaleA);
            if (gr < M && gc + 32 < K)
                __builtin_prefetch(&A[(size_t)gr * lda + gc + 32], 0, 3);
        }
        // ---- stage B tile into [n][k] ----
        if (bT) { // B is N x K row-major
            for (int i = tid; i < 64 * 32; i += 128) {
                int n = i >> 5, c = i & 31;
                int gn = n0 + n, gk = k0 + c;
                float v = (gn < N) ? B[(size_t)gn * ldb + gk] : 0.0f;
                Bsh[n][c] = (_Float16)v;
            }
        } else {  // B is K x N row-major
            for (int i = tid; i < 64 * 32; i += 128) {
                int n = i & 63, c = i >> 6;
                int gn = n0 + n, gk = k0 + c;
                float v = (gn < N) ? B[(size_t)gk * ldb + gn] : 0.0f;
                Bsh[n][c] = (_Float16)v;
            }
        }
        __syncthreads();

        // ---- build 2 A fragments: e<8 -> k=hi*8+e ; e>=8 -> k=16+hi*8+(e-8)
        v16h afrag[2];
        #pragma unroll
        for (int am = 0; am < 2; am++) {
            const _Float16* ap = &As[mrow0 + (am << 4)][hi * 8];
            half8 x0 = *(const half8*)(ap);
            half8 x1 = *(const half8*)(ap + 16);
            #pragma unroll
            for (int j = 0; j < 8; j++) { afrag[am][j] = x0[j]; afrag[am][8 + j] = x1[j]; }
        }
        // ---- 4 B fragments across the N tile, 2 WMMAs each (B reg reuse) ----
        #pragma unroll
        for (int nt = 0; nt < 4; nt++) {
            int ncol = (nt << 4) + (lane & 15);
            const _Float16* bp = &Bsh[ncol][hi * 8];   // e<8 -> k = hi*8 + e ; e>=8 -> k = 16 + hi*8 + (e-8)
            half8 y0 = *(const half8*)(bp);
            half8 y1 = *(const half8*)(bp + 16);
            v16h bfrag;
            #pragma unroll
            for (int j = 0; j < 8; j++) { bfrag[j] = y0[j]; bfrag[8 + j] = y1[j]; }
            acc[0][nt] = __builtin_amdgcn_wmma_f32_16x16x32_f16(
                false, afrag[0], false, bfrag, (short)0, acc[0][nt], false, false);
            acc[1][nt] = __builtin_amdgcn_wmma_f32_16x16x32_f16(
                false, afrag[1], false, bfrag, (short)0, acc[1][nt], false, false);
        }
        __syncthreads();
    }

    // ---- epilogue: D layout: vgpr r, lane -> row = (lane/16)*8 + r, col = lane%16
    const int ncl = lane & 15;
    #pragma unroll
    for (int am = 0; am < 2; am++) {
        const int mbase = m0 + (wave << 5) + (am << 4) + ((lane >> 4) << 3);
        #pragma unroll
        for (int nt = 0; nt < 4; nt++) {
            int col = n0 + (nt << 4) + ncl;
            if (col >= N) continue;
            float bv = bias ? bias[col] : 0.0f;
            #pragma unroll
            for (int r = 0; r < 8; r++) {
                int row = mbase + r;
                if (row >= M) continue;
                float v = acc[am][nt][r] + bv;
                if (resid) v += resid[(size_t)row * ldc + col];
                if (act == 1) { // tanh-approx GELU (jax.nn.gelu default)
                    float u = v;
                    float t = 0.7978845608028654f * (u + 0.044715f * u * u * u);
                    v = 0.5f * u * (1.0f + tanhf(t));
                }
                C[(size_t)row * ldc + col] = v;
            }
        }
    }
}

// =====================================================================
// conv1: (32,1,16,16,16) -> (32,32,8,8,8), k=4 s=2 p=1, relu(+bias)
// =====================================================================
__global__ void conv1_kernel(const float* __restrict__ x, const float* __restrict__ xn,
                             const float* __restrict__ w, const float* __restrict__ b,
                             float* __restrict__ out)
{
    int idx = blockIdx.x * 256 + threadIdx.x;
    if (idx >= 32 * 32 * 8 * 8 * 8) return;
    int ox = idx & 7; int t = idx >> 3;
    int oy = t & 7; t >>= 3;
    int oz = t & 7; t >>= 3;
    int oc = t & 31; t >>= 5;
    int bb = t;
    const float* in = (bb < 16) ? (x + (size_t)bb * 4096) : (xn + (size_t)(bb - 16) * 4096);
    const float* wk = w + (size_t)oc * 64;
    float s = b[oc];
    for (int kz = 0; kz < 4; kz++) {
        int iz = oz * 2 - 1 + kz; if ((unsigned)iz >= 16u) continue;
        for (int ky = 0; ky < 4; ky++) {
            int iy = oy * 2 - 1 + ky; if ((unsigned)iy >= 16u) continue;
            for (int kx = 0; kx < 4; kx++) {
                int ix = ox * 2 - 1 + kx; if ((unsigned)ix >= 16u) continue;
                s += in[(iz * 16 + iy) * 16 + ix] * wk[(kz * 4 + ky) * 4 + kx];
            }
        }
    }
    out[idx] = fmaxf(s, 0.0f);
}

// conv2: (32,32,8,8,8) -> (32,64,4,4,4), k=4 s=2 p=1, +bias
__global__ void conv2_kernel(const float* __restrict__ h1, const float* __restrict__ w,
                             const float* __restrict__ b, float* __restrict__ out)
{
    int idx = blockIdx.x * 256 + threadIdx.x;
    if (idx >= 32 * 64 * 64) return;
    int ox = idx & 3; int t = idx >> 2;
    int oy = t & 3; t >>= 2;
    int oz = t & 3; t >>= 2;
    int oc = t & 63; t >>= 6;
    int bb = t;
    float s = b[oc];
    for (int ic = 0; ic < 32; ic++) {
        const float* in = h1 + (size_t)(bb * 32 + ic) * 512;
        const float* wk = w + (size_t)(oc * 32 + ic) * 64;
        for (int kz = 0; kz < 4; kz++) {
            int iz = oz * 2 - 1 + kz; if ((unsigned)iz >= 8u) continue;
            for (int ky = 0; ky < 4; ky++) {
                int iy = oy * 2 - 1 + ky; if ((unsigned)iy >= 8u) continue;
                for (int kx = 0; kx < 4; kx++) {
                    int ix = ox * 2 - 1 + kx; if ((unsigned)ix >= 8u) continue;
                    s += in[(iz * 8 + iy) * 8 + ix] * wk[(kz * 4 + ky) * 4 + kx];
                }
            }
        }
    }
    out[idx] = s;  // (b,oc,pos)
}

// =====================================================================
// VQ argmin: row r=(b,pos) of z (64 ch) vs 512 codes; d2 = |c|^2 - 2 z.c
// =====================================================================
__global__ __launch_bounds__(64) void vq_kernel(const float* __restrict__ h2,
                                                const float* __restrict__ cb,
                                                int* __restrict__ inds)
{
    int r = blockIdx.x, t = threadIdx.x;
    int bb = r >> 6, pos = r & 63;
    __shared__ float zv[64];
    __shared__ float bdist[64];
    __shared__ int   bidx[64];
    zv[t] = h2[(size_t)(bb * 64 + t) * 64 + pos];
    __syncthreads();
    float best = 3.4e38f; int bi = 0x7fffffff;
    for (int code = t; code < 512; code += 64) {
        const float* c = cb + (size_t)code * 64;
        float dot = 0.f, cc = 0.f;
        for (int j = 0; j < 64; j++) { float cj = c[j]; dot += zv[j] * cj; cc += cj * cj; }
        float d2 = cc - 2.0f * dot;
        if (d2 < best || (d2 == best && code < bi)) { best = d2; bi = code; }
    }
    bdist[t] = best; bidx[t] = bi;
    __syncthreads();
    for (int k = 32; k > 0; k >>= 1) {
        if (t < k) {
            float od = bdist[t + k]; int oi = bidx[t + k];
            if (od < bdist[t] || (od == bdist[t] && oi < bidx[t])) { bdist[t] = od; bidx[t] = oi; }
        }
        __syncthreads();
    }
    if (t == 0) inds[r] = bidx[0];
}

__global__ void build_seq_kernel(const int* __restrict__ iin, const int* __restrict__ inds,
                                 int* __restrict__ seq)
{
    int b = blockIdx.x, l = threadIdx.x; // 16 x 131
    int v;
    if (l < 3)       v = iin[b * 3 + l];
    else if (l < 67) v = inds[(16 + b) * 64 + (l - 3)];   // inds_r
    else             v = inds[b * 64 + (l - 67)];         // inds_i
    seq[b * 131 + l] = v;
}

__global__ void embed_kernel(const int* __restrict__ seq, const float* __restrict__ tok,
                             const float* __restrict__ pos, float* __restrict__ h)
{
    int tkn = blockIdx.x;
    int b = tkn / LSEQ, l = tkn % LSEQ;
    int id = seq[b * 131 + l];
    for (int c = threadIdx.x; c < DIM; c += 128)
        h[(size_t)tkn * DIM + c] = tok[(size_t)id * DIM + c] + pos[(size_t)l * DIM + c];
}

// =====================================================================
// LayerNorm over DIM=512 per token
// =====================================================================
__global__ __launch_bounds__(128) void ln_kernel(const float* __restrict__ x,
                                                 const float* __restrict__ g,
                                                 const float* __restrict__ bta,
                                                 float* __restrict__ y)
{
    int row = blockIdx.x, t = threadIdx.x;
    __shared__ float red[128];
    float s = 0.f, s2 = 0.f;
    for (int c = t; c < DIM; c += 128) { float u = x[(size_t)row * DIM + c]; s += u; s2 += u * u; }
    red[t] = s; __syncthreads();
    for (int k = 64; k > 0; k >>= 1) { if (t < k) red[t] += red[t + k]; __syncthreads(); }
    float mean = red[0] * (1.0f / DIM);
    __syncthreads();
    red[t] = s2; __syncthreads();
    for (int k = 64; k > 0; k >>= 1) { if (t < k) red[t] += red[t + k]; __syncthreads(); }
    float var = red[0] * (1.0f / DIM) - mean * mean;
    float rr = rsqrtf(var + 1e-5f);
    for (int c = t; c < DIM; c += 128) {
        float u = x[(size_t)row * DIM + c];
        y[(size_t)row * DIM + c] = (u - mean) * rr * g[c] + bta[c];
    }
}

// =====================================================================
// Performer feature map.  dd rows are r = (b*L + l)*8 + h, 256 wide.
// diag = 0.5 * sum(dn^2) = 0.0625 * sum(raw^2);   out = exp(dd-diag-mx)/16 + 1e-4
// =====================================================================
__global__ __launch_bounds__(64) void feat_q_kernel(float* __restrict__ dd,
                                                    const float* __restrict__ qraw)
{
    int row = blockIdx.x, t = threadIdx.x;
    __shared__ float red[64];
    float u = qraw[(size_t)row * 64 + t];
    red[t] = u * u; __syncthreads();
    for (int k = 32; k > 0; k >>= 1) { if (t < k) red[t] += red[t + k]; __syncthreads(); }
    float diag = 0.0625f * red[0];
    __syncthreads();
    float m = -3.4e38f;
    for (int c = t; c < MFEAT; c += 64) m = fmaxf(m, dd[(size_t)row * MFEAT + c]);
    red[t] = m; __syncthreads();
    for (int k = 32; k > 0; k >>= 1) { if (t < k) red[t] = fmaxf(red[t], red[t + k]); __syncthreads(); }
    float mx = red[0];
    for (int c = t; c < MFEAT; c += 64) {
        size_t o = (size_t)row * MFEAT + c;
        dd[o] = expf(dd[o] - diag - mx) * 0.0625f + 1e-4f;
    }
}

__global__ __launch_bounds__(64) void rowstat_k_kernel(const float* __restrict__ dd,
                                                       const float* __restrict__ kraw,
                                                       float* __restrict__ diag,
                                                       float* __restrict__ rowmax)
{
    int row = blockIdx.x, t = threadIdx.x;
    __shared__ float red[64];
    float u = kraw[(size_t)row * 64 + t];
    red[t] = u * u; __syncthreads();
    for (int k = 32; k > 0; k >>= 1) { if (t < k) red[t] += red[t + k]; __syncthreads(); }
    if (t == 0) diag[row] = 0.0625f * red[0];
    __syncthreads();
    float m = -3.4e38f;
    for (int c = t; c < MFEAT; c += 64) m = fmaxf(m, dd[(size_t)row * MFEAT + c]);
    red[t] = m; __syncthreads();
    for (int k = 32; k > 0; k >>= 1) { if (t < k) red[t] = fmaxf(red[t], red[t + k]); __syncthreads(); }
    if (t == 0) rowmax[row] = red[0];
}

__global__ void gmax_kernel(const float* __restrict__ rowmax, int n, float* __restrict__ out)
{
    __shared__ float red[256];
    int t = threadIdx.x;
    float m = -3.4e38f;
    for (int j = t; j < n; j += 256) m = fmaxf(m, rowmax[j]);
    red[t] = m; __syncthreads();
    for (int k = 128; k > 0; k >>= 1) { if (t < k) red[t] = fmaxf(red[t], red[t + k]); __syncthreads(); }
    if (t == 0) out[0] = red[0];
}

__global__ void feat_k_apply_kernel(float* __restrict__ dd, const float* __restrict__ diag,
                                    const float* __restrict__ gmx, int total)
{
    int idx = blockIdx.x * 256 + threadIdx.x;
    if (idx >= total) return;
    int row = idx >> 8;
    dd[idx] = expf(dd[idx] - diag[row] - gmx[0]) * 0.0625f + 1e-4f;
}

// =====================================================================
// den[b,h,l] = qp_l . cumsum(kp)_l   (inclusive)
// =====================================================================
__global__ __launch_bounds__(64) void attn_den_kernel(const float* __restrict__ kp,
                                                      const float* __restrict__ qp,
                                                      float* __restrict__ den)
{
    int bh = blockIdx.x, t = threadIdx.x;
    int b = bh >> 3, h = bh & 7;
    __shared__ float z[256];
    for (int j = t; j < 256; j += 64) z[j] = 0.f;
    __syncthreads();
    for (int l = 0; l < LSEQ; l++) {
        size_t roff = ((size_t)((b * LSEQ + l) * NHEADS + h)) * MFEAT;
        for (int j = t; j < 256; j += 64) z[j] += kp[roff + j];
        __syncthreads();
        float s = 0.f;
        for (int m = 0; m < 256; m++) s += qp[roff + m] * z[m];
        if (t == 0) den[bh * LSEQ + l] = s;
        __syncthreads();
    }
}

// =====================================================================
// Causal linear attention, streaming inclusive scan.
// Thread d keeps S[64] (one m-chunk column) in registers; 4 chunks over M.
// Writes o directly in (token, h*64+d) layout; divides by den on last chunk.
// =====================================================================
__global__ __launch_bounds__(64) void attn_accum_kernel(const float* __restrict__ kp,
                                                        const float* __restrict__ qp,
                                                        const float* __restrict__ v,
                                                        const float* __restrict__ den,
                                                        float* __restrict__ o_tok)
{
    int bh = blockIdx.x;
    int b = bh >> 3, h = bh & 7;
    int d = threadIdx.x;
    for (int mc = 0; mc < 4; mc++) {
        float S[64];
        #pragma unroll
        for (int j = 0; j < 64; j++) S[j] = 0.f;
        int m0 = mc * 64;
        for (int l = 0; l < LSEQ; l++) {
            size_t roff = ((size_t)((b * LSEQ + l) * NHEADS + h)) * MFEAT + m0;
            float vd = v[(size_t)(b * LSEQ + l) * DIM + h * DH + d];
            float accum = 0.f;
            #pragma unroll
            for (int j = 0; j < 64; j++) {
                float s = S[j] + kp[roff + j] * vd;
                S[j] = s;
                accum += qp[roff + j] * s;
            }
            float* op = &o_tok[(size_t)(b * LSEQ + l) * DIM + h * DH + d];
            float prev = (mc == 0) ? 0.f : *op;
            float nv = prev + accum;
            if (mc == 3) nv /= den[bh * LSEQ + l];
            *op = nv;
        }
    }
}

// =====================================================================
// Loss: per token row of logits (523), lse - logit[tgt], mean via atomicAdd
// =====================================================================
__global__ void zero_out_kernel(float* __restrict__ out) { out[0] = 0.0f; }

__global__ __launch_bounds__(64) void loss_kernel(const float* __restrict__ logits,
                                                  const int* __restrict__ seq,
                                                  float* __restrict__ out)
{
    int tkn = blockIdx.x, t = threadIdx.x;
    int b = tkn / LSEQ, l = tkn % LSEQ;
    int tgt = seq[b * 131 + l + 1];
    const float* lg = logits + (size_t)tkn * VOCAB;
    __shared__ float red[64];
    float m = -3.4e38f;
    for (int c = t; c < VOCAB; c += 64) m = fmaxf(m, lg[c]);
    red[t] = m; __syncthreads();
    for (int k = 32; k > 0; k >>= 1) { if (t < k) red[t] = fmaxf(red[t], red[t + k]); __syncthreads(); }
    float mx = red[0];
    __syncthreads();
    float s = 0.f;
    for (int c = t; c < VOCAB; c += 64) s += expf(lg[c] - mx);
    red[t] = s; __syncthreads();
    for (int k = 32; k > 0; k >>= 1) { if (t < k) red[t] += red[t + k]; __syncthreads(); }
    if (t == 0) {
        float lse = mx + logf(red[0]);
        float nll = lse - lg[tgt];
        atomicAdd(out, nll * (1.0f / NTOK));
    }
}

// =====================================================================
// host orchestration
// =====================================================================
static inline void launch_gemm(const float* A, int lda, const float* B, int ldb,
                               const float* bias, const float* resid,
                               float* C, int ldc, int M, int N, int K,
                               float scaleA, int bT, int act, hipStream_t s)
{
    dim3 grid((N + 63) / 64, (M + 127) / 128);
    gemm_wmma_kernel<<<grid, 128, 0, s>>>(A, lda, B, ldb, bias, resid, C, ldc,
                                          M, N, K, scaleA, bT, act);
}

extern "C" void kernel_launch(void* const* d_in, const int* in_sizes, int n_in,
                              void* d_out, int out_size, void* d_ws, size_t ws_size,
                              hipStream_t stream)
{
    const float* x        = (const float*)d_in[0];
    const float* xn       = (const float*)d_in[1];
    const int*   iin      = (const int*)  d_in[2];
    const float* conv1_w  = (const float*)d_in[3];
    const float* conv1_b  = (const float*)d_in[4];
    const float* conv2_w  = (const float*)d_in[5];
    const float* conv2_b  = (const float*)d_in[6];
    const float* codebook = (const float*)d_in[7];
    const float* proj     = (const float*)d_in[8];
    const float* tok_emb  = (const float*)d_in[9];
    const float* pos_emb  = (const float*)d_in[10];
    const float* ln1_g    = (const float*)d_in[11];
    const float* ln1_b    = (const float*)d_in[12];
    const float* wq       = (const float*)d_in[13];
    const float* wk       = (const float*)d_in[14];
    const float* wv       = (const float*)d_in[15];
    const float* wo       = (const float*)d_in[16];
    const float* bo       = (const float*)d_in[17];
    const float* ln2_g    = (const float*)d_in[18];
    const float* ln2_b    = (const float*)d_in[19];
    const float* ff1_w    = (const float*)d_in[20];
    const float* ff1_b    = (const float*)d_in[21];
    const float* ff2_w    = (const float*)d_in[22];
    const float* ff2_b    = (const float*)d_in[23];
    const float* lnf_g    = (const float*)d_in[24];
    const float* lnf_b    = (const float*)d_in[25];
    const float* logits_w = (const float*)d_in[26];
    const float* logits_b = (const float*)d_in[27];

    float* ws = (float*)d_ws;
    size_t off = 0;
    auto alloc = [&](size_t n) { float* p = ws + off; off += n; return p; };
    float* hbuf  = alloc((size_t)NTOK * DIM);
    float* x1    = alloc((size_t)NTOK * DIM);
    float* qb    = alloc((size_t)NTOK * DIM);
    float* kb    = alloc((size_t)NTOK * DIM);
    float* vb    = alloc((size_t)NTOK * DIM);
    float* otok  = alloc((size_t)NTOK * DIM);
    float* qp    = alloc((size_t)NROWS * MFEAT);   // also conv bufs, then FFN mid
    float* kpb   = alloc((size_t)NROWS * MFEAT);   // also logits buffer
    float* den   = alloc(NROWS);
    float* diagk = alloc(NROWS);
    float* rmaxk = alloc(NROWS);
    float* gmx   = alloc(16);
    int* inds = (int*)alloc(2048);
    int* seq  = (int*)alloc(2112);
    (void)ws_size; (void)n_in; (void)in_sizes; (void)out_size;

    float* h1c = qp;                  // conv1 out: 32*32*512 = 524288 floats
    float* h2c = qp + 524288;         // conv2 out: 32*64*64  = 131072 floats
    float* mid = qp;                  // FFN hidden (2080x2048) reuses qp region
    float* logitsb = kpb;             // logits (2080x523) reuse kp region
    float* dout = (float*)d_out;

    // ---------------- VQ encode + sequence build ----------------
    conv1_kernel<<<(32 * 32 * 512 + 255) / 256, 256, 0, stream>>>(x, xn, conv1_w, conv1_b, h1c);
    conv2_kernel<<<(32 * 64 * 64 + 255) / 256, 256, 0, stream>>>(h1c, conv2_w, conv2_b, h2c);
    vq_kernel<<<2048, 64, 0, stream>>>(h2c, codebook, inds);
    build_seq_kernel<<<16, 131, 0, stream>>>(iin, inds, seq);
    embed_kernel<<<NTOK, 128, 0, stream>>>(seq, tok_emb, pos_emb, hbuf);

    const float dn_scale = 0.35355339059327373f;  // 64^-0.25

    // ---------------- transformer layers ----------------
    for (int l = 0; l < DEPTH; l++) {
        const float* Wq = wq + (size_t)l * DIM * DIM;
        const float* Wk = wk + (size_t)l * DIM * DIM;
        const float* Wv = wv + (size_t)l * DIM * DIM;
        const float* Wo = wo + (size_t)l * DIM * DIM;
        const float* Pl = proj + (size_t)l * MFEAT * DH;

        ln_kernel<<<NTOK, 128, 0, stream>>>(hbuf, ln1_g + l * DIM, ln1_b + l * DIM, x1);
        launch_gemm(x1, DIM, Wq, DIM, nullptr, nullptr, qb, DIM, NTOK, DIM, DIM, 1.f, 0, 0, stream);
        launch_gemm(x1, DIM, Wk, DIM, nullptr, nullptr, kb, DIM, NTOK, DIM, DIM, 1.f, 0, 0, stream);
        launch_gemm(x1, DIM, Wv, DIM, nullptr, nullptr, vb, DIM, NTOK, DIM, DIM, 1.f, 0, 0, stream);

        // dd = (q * 64^-0.25) @ proj^T   rows=(b,l,h), B is N x K (bT)
        launch_gemm(qb, DH, Pl, DH, nullptr, nullptr, qp,  MFEAT, NROWS, MFEAT, DH, dn_scale, 1, 0, stream);
        launch_gemm(kb, DH, Pl, DH, nullptr, nullptr, kpb, MFEAT, NROWS, MFEAT, DH, dn_scale, 1, 0, stream);

        feat_q_kernel<<<NROWS, 64, 0, stream>>>(qp, qb);
        rowstat_k_kernel<<<NROWS, 64, 0, stream>>>(kpb, kb, diagk, rmaxk);
        gmax_kernel<<<1, 256, 0, stream>>>(rmaxk, NROWS, gmx);
        feat_k_apply_kernel<<<(NROWS * MFEAT + 255) / 256, 256, 0, stream>>>(kpb, diagk, gmx, NROWS * MFEAT);

        attn_den_kernel<<<BS * NHEADS, 64, 0, stream>>>(kpb, qp, den);
        attn_accum_kernel<<<BS * NHEADS, 64, 0, stream>>>(kpb, qp, vb, den, otok);

        // h = h + o @ wo + bo
        launch_gemm(otok, DIM, Wo, DIM, bo + l * DIM, hbuf, hbuf, DIM, NTOK, DIM, DIM, 1.f, 0, 0, stream);

        // FFN
        ln_kernel<<<NTOK, 128, 0, stream>>>(hbuf, ln2_g + l * DIM, ln2_b + l * DIM, x1);
        launch_gemm(x1, DIM, ff1_w + (size_t)l * DIM * 4 * DIM, 4 * DIM,
                    ff1_b + l * 4 * DIM, nullptr, mid, 4 * DIM, NTOK, 4 * DIM, DIM, 1.f, 0, 1, stream);
        launch_gemm(mid, 4 * DIM, ff2_w + (size_t)l * 4 * DIM * DIM, DIM,
                    ff2_b + l * DIM, hbuf, hbuf, DIM, NTOK, DIM, 4 * DIM, 1.f, 0, 0, stream);
    }

    // ---------------- head + loss ----------------
    ln_kernel<<<NTOK, 128, 0, stream>>>(hbuf, lnf_g, lnf_b, x1);
    launch_gemm(x1, DIM, logits_w, VOCAB, logits_b, nullptr, logitsb, VOCAB,
                NTOK, VOCAB, DIM, 1.f, 0, 0, stream);
    zero_out_kernel<<<1, 1, 0, stream>>>(dout);
    loss_kernel<<<NTOK, 64, 0, stream>>>(logitsb, seq, dout);
}
